// ParallelSubPolicies_6116033429645
// MI455X (gfx1250) — matrix operand order repaired and checked by reading, not compiled, not run on hardware
//
#include <hip/hip_runtime.h>
#include <stdint.h>

#define OBS_D 128
#define HID_D 256
#define ACT_D 32
#define P_N   64
#define B_N   4096
#define MT    128          // batch rows per block
#define TPB   256          // 8 waves (wave32)
#define EPSV  1e-5f

#define ACT_PITCH 264      // 256 + 8 halves pad -> conflict-free b128 A-frag loads
#define W_PITCH   40       // 32 + 8 halves pad  -> 16B-aligned, conflict-free B-frag loads

// ---- preprocessed-weight workspace layout (bf16 halves) ----
// chunk = 32 K-values x DOUT, stored transposed+padded: [n][40]
#define CH_BIG   (HID_D * W_PITCH)              // 10240 halves = 20480 B (DOUT=256)
#define CH_SMALL 2048                            // padded to 4096 B      (DOUT=32)
#define KC1 (OBS_D / 32)                         // 4
#define KCH (HID_D / 32)                         // 8
#define WS_L1_HALVES ((size_t)P_N * KC1 * CH_BIG)
#define WS_L2_HALVES ((size_t)P_N * KCH * CH_BIG)
#define WS_L4_HALVES ((size_t)P_N * KCH * CH_SMALL)
#define WS_O1 ((size_t)0)
#define WS_O2 (WS_O1 + WS_L1_HALVES)
#define WS_O3 (WS_O2 + WS_L2_HALVES)
#define WS_O4 (WS_O3 + WS_L2_HALVES)
#define WS_TOTAL_HALVES (WS_O4 + WS_L4_HALVES)
#define WS_TOTAL_BYTES  (WS_TOTAL_HALVES * 2)    // ~27 MB

#define WBUF_HALVES CH_BIG                       // one staging buffer (20 KB)

typedef __attribute__((ext_vector_type(8)))  __bf16 bf16x8;
typedef __attribute__((ext_vector_type(16))) __bf16 bf16x16;
typedef __attribute__((ext_vector_type(8)))  float  f32x8;

union BF16V {
  bf16x16 v;
  struct { bf16x8 lo, hi; } h;
};

// ---------------- CDNA5 async copy helpers (inline asm; tracked by ASYNCcnt) ---------
__device__ __forceinline__ void async_b128(unsigned lds_addr, const void* gaddr) {
  asm volatile("global_load_async_to_lds_b128 %0, %1, off"
               :: "v"(lds_addr), "v"(gaddr) : "memory");
}
template<int N>
__device__ __forceinline__ void wait_async() {
  asm volatile("s_wait_asynccnt %0" :: "n"(N) : "memory");
}

// ---------------- one-shot weight preprocessing: f32 [P][DIN][DOUT] ->
// bf16 chunk-transposed [p][kc][n][W_PITCH] (pad halves unread) ---------------------
template<int DIN, int DOUT, int CH>
__global__ __launch_bounds__(TPB) void preproc_weights_kernel(
    const float* __restrict__ W, __bf16* __restrict__ dst)
{
  constexpr int KC = DIN / 32;
  const int TOT = P_N * DIN * DOUT;
  for (int i = blockIdx.x * TPB + threadIdx.x; i < TOT; i += gridDim.x * TPB) {
    int n  = i % DOUT;
    int k  = (i / DOUT) % DIN;
    int p  = i / (DIN * DOUT);
    int kc = k >> 5;
    int kk = k & 31;
    dst[(size_t)(p * KC + kc) * CH + n * W_PITCH + kk] = (__bf16)W[i];
  }
}

// ---------------- fused layer: GEMM (WMMA bf16) + bias + GroupNorm + ELU ------------
template<int DIN, int DOUT, bool DOELU, bool LAST, bool ASYNCW, int CH>
__device__ __forceinline__ void layer_gemm(
    const float* __restrict__ W, const __bf16* __restrict__ Wws,
    const float* __restrict__ bias,
    const float* __restrict__ gam, const float* __restrict__ bet,
    float* __restrict__ out, int p, int b0,
    __bf16 (*s_act)[ACT_PITCH], __bf16* s_w /* 2 * WBUF_HALVES */)
{
  constexpr int NT   = DOUT / 16;     // n-tiles per wave (wave owns full DOUT width)
  constexpr int KC   = DIN / 32;      // K chunks
  constexpr int CHB  = CH * 2;        // chunk bytes in ws
  constexpr int PERW = CHB / (TPB * 16); // async b128 ops per thread per chunk
  static_assert(!ASYNCW || CHB % (TPB * 16) == 0, "chunk must tile by 256x16B");

  const int tid   = threadIdx.x;
  const int lane  = tid & 31;
  const int wid   = tid >> 5;
  const int l16   = lane & 15;
  const int koffA = (lane < 16) ? 0 : 8;    // ISA 16-bit A 16x32 layout
  const int koffB = (lane < 16) ? 0 : 8;    // ISA 16-bit B 32x16 layout
  const int arow  = wid * 16 + l16;

  f32x8 acc[NT];
#pragma unroll
  for (int t = 0; t < NT; ++t)
    acc[t] = (f32x8){0.f, 0.f, 0.f, 0.f, 0.f, 0.f, 0.f, 0.f};

  auto stage = [&](int kc, int bufi) {
    if constexpr (ASYNCW) {
      const char* g = (const char*)(Wws + (size_t)(p * KC + kc) * CH);
      unsigned lbase = (unsigned)(uintptr_t)(s_w + bufi * WBUF_HALVES);
#pragma unroll
      for (int i = 0; i < PERW; ++i)
        async_b128(lbase + tid * 16 + i * (TPB * 16),
                   g + tid * 16 + i * (TPB * 16));
    } else {
      const float* wsrc = W + ((size_t)p * DIN + (size_t)kc * 32) * DOUT;
      __bf16* dstb = s_w + bufi * WBUF_HALVES;
      for (int i = tid; i < 32 * DOUT; i += TPB) {
        int kk = i / DOUT;
        int n  = i - kk * DOUT;
        dstb[n * W_PITCH + kk] = (__bf16)wsrc[kk * DOUT + n];
      }
    }
  };

  stage(0, 0);
  for (int kc = 0; kc < KC; ++kc) {
    const int cur = kc & 1;
    if (kc + 1 < KC) stage(kc + 1, cur ^ 1);
    if constexpr (ASYNCW) {
      if (kc + 1 < KC) wait_async<PERW>();  // current chunk landed; next may fly
      else             wait_async<0>();
    }
    __syncthreads();

    const __bf16* wb = s_w + cur * WBUF_HALVES;

    // A fragment (16x32 bf16): halves 0..7 = K koffA..+7, halves 8..15 = K koffA+16..+23
    BF16V af;
    af.h.lo = *(const bf16x8*)&s_act[arow][kc * 32 + koffA];
    af.h.hi = *(const bf16x8*)&s_act[arow][kc * 32 + koffA + 16];

    // software-pipelined B fragments: prefetch t+1 while WMMA t issues
    BF16V bcur;
    bcur.h.lo = *(const bf16x8*)&wb[l16 * W_PITCH + koffB];
    bcur.h.hi = *(const bf16x8*)&wb[l16 * W_PITCH + koffB + 16];
#pragma unroll
    for (int t = 0; t < NT; ++t) {
      BF16V bnext = bcur;
      if (t + 1 < NT) {
        bnext.h.lo = *(const bf16x8*)&wb[((t + 1) * 16 + l16) * W_PITCH + koffB];
        bnext.h.hi = *(const bf16x8*)&wb[((t + 1) * 16 + l16) * W_PITCH + koffB + 16];
      }
      acc[t] = __builtin_amdgcn_wmma_f32_16x16x32_bf16(
          false, af.v, false, bcur.v, (short)0, acc[t], false, false);
      bcur = bnext;
    }
    __syncthreads();
  }

  // Epilogue: +bias, GroupNorm over DOUT per row, (ELU), writeback
  float bv[NT], gv[NT], ev[NT];
#pragma unroll
  for (int t = 0; t < NT; ++t) {
    int n = t * 16 + l16;
    bv[t] = bias[p * DOUT + n];
    gv[t] = gam [p * DOUT + n];
    ev[t] = bet [p * DOUT + n];
  }

#pragma unroll
  for (int s = 0; s < 8; ++s) {
    // slot s: lanes 0-15 hold row (wid*16+s), lanes 16-31 hold row (wid*16+s+8)
    float s1 = 0.f, s2 = 0.f;
#pragma unroll
    for (int t = 0; t < NT; ++t) {
      float v = acc[t][s] + bv[t];
      acc[t][s] = v;
      s1 += v;
      s2 += v * v;
    }
#pragma unroll
    for (int m = 1; m < 16; m <<= 1) {  // butterfly within each 16-lane half
      s1 += __shfl_xor(s1, m, 32);
      s2 += __shfl_xor(s2, m, 32);
    }
    float mu  = s1 * (1.f / DOUT);
    float var = s2 * (1.f / DOUT) - mu * mu;
    float sc  = __frsqrt_rn(var + EPSV);
    int row   = wid * 16 + s + ((lane >> 4) << 3);
#pragma unroll
    for (int t = 0; t < NT; ++t) {
      float v = (acc[t][s] - mu) * sc * gv[t] + ev[t];
      if (DOELU) v = (v > 0.f) ? v : (__expf(v) - 1.f);
      if (LAST) {
        out[((size_t)(b0 + row) * P_N + p) * ACT_D + t * 16 + l16] = v;
      } else {
        s_act[row][t * 16 + l16] = (__bf16)v;   // own-wave rows only: no race
      }
    }
  }
}

template<bool ASYNCW>
__global__ __launch_bounds__(TPB) void subpolicies_fused_kernel(
    const float* __restrict__ obs,
    const float* __restrict__ W1, const float* __restrict__ b1,
    const float* __restrict__ g1, const float* __restrict__ be1,
    const float* __restrict__ W2, const float* __restrict__ b2,
    const float* __restrict__ g2, const float* __restrict__ be2,
    const float* __restrict__ W3, const float* __restrict__ b3,
    const float* __restrict__ g3, const float* __restrict__ be3,
    const float* __restrict__ W4, const float* __restrict__ b4,
    const float* __restrict__ g4, const float* __restrict__ be4,
    const __bf16* __restrict__ ws,
    float* __restrict__ out)
{
  __shared__ __align__(16) __bf16 s_act[MT][ACT_PITCH];   // ~66 KB
  __shared__ __align__(16) __bf16 s_w[2 * WBUF_HALVES];   // ~40 KB double-buffered

  const int p  = blockIdx.y;
  const int b0 = blockIdx.x * MT;

  // Stage observation tile (f32 -> bf16): broadcast input for this sub-policy
  for (int i = threadIdx.x; i < MT * OBS_D; i += TPB) {
    int r = i >> 7;
    int c = i & 127;
    s_act[r][c] = (__bf16)obs[(size_t)(b0 + r) * OBS_D + c];
  }
  __syncthreads();

  const __bf16* ws1 = ws + WS_O1;
  const __bf16* ws2 = ws + WS_O2;
  const __bf16* ws3 = ws + WS_O3;
  const __bf16* ws4 = ws + WS_O4;

  layer_gemm<OBS_D, HID_D, true,  false, ASYNCW, CH_BIG  >(W1, ws1, b1, g1, be1, nullptr, p, b0, s_act, s_w);
  layer_gemm<HID_D, HID_D, true,  false, ASYNCW, CH_BIG  >(W2, ws2, b2, g2, be2, nullptr, p, b0, s_act, s_w);
  layer_gemm<HID_D, HID_D, true,  false, ASYNCW, CH_BIG  >(W3, ws3, b3, g3, be3, nullptr, p, b0, s_act, s_w);
  layer_gemm<HID_D, ACT_D, false, true,  ASYNCW, CH_SMALL>(W4, ws4, b4, g4, be4, out,     p, b0, s_act, s_w);
}

extern "C" void kernel_launch(void* const* d_in, const int* in_sizes, int n_in,
                              void* d_out, int out_size, void* d_ws, size_t ws_size,
                              hipStream_t stream) {
  (void)in_sizes; (void)n_in; (void)out_size;
  const float* obs = (const float*)d_in[0];
  const float* W1  = (const float*)d_in[1];
  const float* b1  = (const float*)d_in[2];
  const float* g1  = (const float*)d_in[3];
  const float* be1 = (const float*)d_in[4];
  const float* W2  = (const float*)d_in[5];
  const float* b2  = (const float*)d_in[6];
  const float* g2  = (const float*)d_in[7];
  const float* be2 = (const float*)d_in[8];
  const float* W3  = (const float*)d_in[9];
  const float* b3  = (const float*)d_in[10];
  const float* g3  = (const float*)d_in[11];
  const float* be3 = (const float*)d_in[12];
  const float* W4  = (const float*)d_in[13];
  const float* b4  = (const float*)d_in[14];
  const float* g4  = (const float*)d_in[15];
  const float* g4e = (const float*)d_in[16];
  float* out = (float*)d_out;

  const bool use_ws = (d_ws != nullptr) && (ws_size >= WS_TOTAL_BYTES);
  dim3 grid(B_N / MT, P_N);   // (32, 64)

  if (use_ws) {
    __bf16* ws = (__bf16*)d_ws;
    // One-shot (per call) bf16 conversion + chunk-transpose of all weights.
    preproc_weights_kernel<OBS_D, HID_D, CH_BIG  ><<<1024, TPB, 0, stream>>>(W1, ws + WS_O1);
    preproc_weights_kernel<HID_D, HID_D, CH_BIG  ><<<2048, TPB, 0, stream>>>(W2, ws + WS_O2);
    preproc_weights_kernel<HID_D, HID_D, CH_BIG  ><<<2048, TPB, 0, stream>>>(W3, ws + WS_O3);
    preproc_weights_kernel<HID_D, ACT_D, CH_SMALL><<< 512, TPB, 0, stream>>>(W4, ws + WS_O4);
    subpolicies_fused_kernel<true><<<grid, TPB, 0, stream>>>(
        obs, W1, b1, g1, be1, W2, b2, g2, be2,
        W3, b3, g3, be3, W4, b4, g4, g4e, ws, out);
  } else {
    subpolicies_fused_kernel<false><<<grid, TPB, 0, stream>>>(
        obs, W1, b1, g1, be1, W2, b2, g2, be2,
        W3, b3, g3, be3, W4, b4, g4, g4e, nullptr, out);
  }
}
